// DynamicKGating_37005438223071
// MI455X (gfx1250) — matrix operand order repaired and checked by reading, not compiled, not run on hardware
//
#include <hip/hip_runtime.h>
#include <hip/hip_bf16.h>

typedef __attribute__((ext_vector_type(16))) _Float16 v16h;
typedef __attribute__((ext_vector_type(8)))  float    v8f;

namespace {
constexpr int   kB = 8, kT = 2048, kD = 1024, kE = 64, kC = 40;
constexpr int   kTok = kB * kT;                       // 16384 tokens
constexpr float kThresh = 0.8f;
constexpr long long kDispatchElems = (long long)kB * kT * kE * kC;  // 41,943,040
}

// ---------------------------------------------------------------------------
// 1) Wide zero-fill of the output region (dominant memory cost: 335 MB).
__global__ void zero_f4(float4* __restrict__ p, long long n4) {
    long long i = (long long)blockIdx.x * blockDim.x + threadIdx.x;
    long long stride = (long long)gridDim.x * blockDim.x;
    float4 z; z.x = 0.f; z.y = 0.f; z.z = 0.f; z.w = 0.f;
    for (; i < n4; i += stride) p[i] = z;
}

__global__ void zero_f(float* __restrict__ p, int n) {
    int i = blockIdx.x * blockDim.x + threadIdx.x;
    if (i < n) p[i] = 0.f;
}

// ---------------------------------------------------------------------------
// 2) Swizzle w_gating [D,E] f32 -> f16 B-fragments laid out exactly as the
//    16-bit 32x16 WMMA B operand: per (ntile, kstep, lane) a contiguous run of
//    16 halfs, element i = W[kstep*32 + (lane<16?0:8) + (i&7) + (i>=8?16:0)][ntile*16 + lane%16].
__global__ void wconv(const float* __restrict__ w, _Float16* __restrict__ wB) {
    int idx  = blockIdx.x * blockDim.x + threadIdx.x;   // 65536 total
    int i    = idx & 15;
    int lane = (idx >> 4) & 31;
    int ks   = (idx >> 9) & 31;
    int nt   = idx >> 14;
    int n    = nt * 16 + (lane & 15);
    int k    = ks * 32 + ((lane < 16) ? 0 : 8) + (i & 7) + ((i >> 3) << 4);
    wB[idx]  = (_Float16)w[k * kE + n];
}

// ---------------------------------------------------------------------------
// 3) Gating: WMMA GEMM (16 tok x 64 experts per wave) fused with softmax,
//    threshold selection, renorm. Writes dense weights [B*T, E] to ws and
//    accumulates per-(b,e) prob/mask sums for the aux loss.
__global__ __launch_bounds__(256) void gating(const float* __restrict__ x,
                                              const _Float16* __restrict__ wB,
                                              float* __restrict__ weights,
                                              float* __restrict__ probs_sum,
                                              float* __restrict__ mask_sum) {
    __shared__ float lds_p[128][kE + 1];   // +1 pad: conflict-free rows
    __shared__ float probs_part[kE];
    __shared__ float mask_part[kE];

    const int tid = threadIdx.x;
    if (tid < kE) { probs_part[tid] = 0.f; mask_part[tid] = 0.f; }

    const int wave = tid >> 5;
    const int lane = tid & 31;
    const int m    = lane & 15;       // row within 16-token tile
    const int hf   = lane >> 4;       // half-wave: selects K sub-block
    const int tokBase = blockIdx.x * 128 + wave * 16;

    // A operand: row (tokBase+m); lanes 0-15 carry K {0..7,16..23}, 16-31 {8..15,24..31}
    const float* xrow = x + (size_t)(tokBase + m) * kD + hf * 8;

    v8f acc[4];
#pragma unroll
    for (int i = 0; i < 4; ++i) acc[i] = v8f{};

    for (int ks = 0; ks < 32; ++ks) {
        const float* xp = xrow + ks * 32;
        v16h a;
#pragma unroll
        for (int i = 0; i < 8; ++i) {
            a[i]     = (_Float16)xp[i];        // K = off + 0..7
            a[8 + i] = (_Float16)xp[16 + i];   // K = off + 16..23
        }
#pragma unroll
        for (int nt = 0; nt < 4; ++nt) {
            v16h bv = *((const v16h*)wB + ((nt * 32 + ks) * 32 + lane));
            acc[nt] = __builtin_amdgcn_wmma_f32_16x16x32_f16(
                false, a, false, bv, (short)0, acc[nt], false, false);
        }
    }

    // C layout: VGPR r, lanes 0-15 -> (M=r, N=lane), lanes 16-31 -> (M=r+8, N=lane-16)
#pragma unroll
    for (int nt = 0; nt < 4; ++nt)
#pragma unroll
        for (int r = 0; r < 8; ++r)
            lds_p[wave * 16 + hf * 8 + r][nt * 16 + m] = acc[nt][r];
    __syncthreads();

    // Phase 2: one thread per token (128 tokens/block).
    if (tid < 128) {
        float* pl = lds_p[tid];
        float mx = pl[0];
        for (int j = 1; j < kE; ++j) mx = fmaxf(mx, pl[j]);
        float sum = 0.f;
        for (int j = 0; j < kE; ++j) { float v = __expf(pl[j] - mx); pl[j] = v; sum += v; }
        float inv = 1.f / sum;
        for (int j = 0; j < kE; ++j) pl[j] *= inv;   // probs now in LDS

        // Selection: expert e kept iff (sum of probs ranked above it) + p_e < T,
        // or it is the argmax (S==0). Stable-sort tie-break by index.
        unsigned long long keep = 0ull;
        float renorm = 0.f;
        for (int e = 0; e < kE; ++e) {
            float pe = pl[e];
            float S = 0.f;
            for (int j = 0; j < kE; ++j) {
                float pj = pl[j];
                if (pj > pe || (pj == pe && j < e)) S += pj;
            }
            if (S + pe < kThresh || S == 0.f) { keep |= (1ull << e); renorm += pe; }
        }
        float invr = 1.f / fmaxf(renorm, 1e-9f);

        int g = blockIdx.x * 128 + tid;
        float* wout = weights + (size_t)g * kE;
        for (int ii = 0; ii < kE; ++ii) {
            int e = (ii + tid) & (kE - 1);   // stagger to spread LDS atomic banks
            float pe = pl[e];
            bool  kk = (keep >> e) & 1ull;
            wout[e] = kk ? pe * invr : 0.f;
            atomicAdd(&probs_part[e], pe);
            if (kk) atomicAdd(&mask_part[e], 1.f);
        }
    }
    __syncthreads();
    if (tid < kE) {
        int b = (blockIdx.x * 128) / kT;     // blocks never straddle a batch
        atomicAdd(&probs_sum[b * kE + tid], probs_part[tid]);
        atomicAdd(&mask_sum[b * kE + tid], mask_part[tid]);
    }
}

// ---------------------------------------------------------------------------
// 4) Per-(b,e) capacity scan + sparse scatter. One wave32 per (b,e); each lane
//    owns 64 consecutive tokens; wave-wide exclusive prefix via __shfl_up.
__global__ __launch_bounds__(32) void scan_scatter(const float* __restrict__ weights,
                                                   float* __restrict__ dispatch,
                                                   float* __restrict__ combine) {
    const int be = blockIdx.x;
    const int b  = be >> 6;
    const int e  = be & 63;
    const int lane = threadIdx.x;
    const float* wcol = weights + (size_t)b * kT * kE + e;
    const int t0 = lane * 64;

    int c = 0;
    for (int i = 0; i < 64; ++i)
        c += (wcol[(size_t)(t0 + i) * kE] > 0.f) ? 1 : 0;

    int pre = c;
    for (int d = 1; d < 32; d <<= 1) {       // wave32 inclusive scan
        int v = __shfl_up(pre, d, 32);
        if (lane >= d) pre += v;
    }
    int pos = pre - c;                        // exclusive prefix

    for (int i = 0; i < 64; ++i) {
        int t = t0 + i;
        float wv = wcol[(size_t)t * kE];
        if (wv > 0.f) {
            if (pos < kC) {
                size_t o = (((size_t)(b * kT + t) * kE) + e) * kC + pos;
                dispatch[o] = 1.0f;
                combine[o]  = wv;
            }
            ++pos;
        }
    }
}

// ---------------------------------------------------------------------------
// 5) aux = mean_{b,e}(density * density_proxy) * E^2 = sum * E / B
__global__ __launch_bounds__(512) void aux_kernel(const float* __restrict__ probs_sum,
                                                  const float* __restrict__ mask_sum,
                                                  float* __restrict__ out_aux) {
    __shared__ float red[512];
    int tid = threadIdx.x;
    float v = (mask_sum[tid] * (1.f / kT)) * (probs_sum[tid] * (1.f / kT));
    red[tid] = v;
    __syncthreads();
    for (int s = 256; s > 0; s >>= 1) {
        if (tid < s) red[tid] += red[tid + s];
        __syncthreads();
    }
    if (tid == 0) *out_aux = red[0] * ((float)kE / (float)kB);
}

// ---------------------------------------------------------------------------
extern "C" void kernel_launch(void* const* d_in, const int* in_sizes, int n_in,
                              void* d_out, int out_size, void* d_ws, size_t ws_size,
                              hipStream_t stream) {
    (void)in_sizes; (void)n_in; (void)out_size; (void)ws_size;
    const float* x  = (const float*)d_in[0];
    const float* wg = (const float*)d_in[1];
    float* out = (float*)d_out;

    char* ws = (char*)d_ws;
    float*    weights   = (float*)ws;                              // 4 MiB
    _Float16* wB        = (_Float16*)(ws + (size_t)kTok * kE * 4); // 128 KiB
    float*    probs_sum = (float*)(ws + (size_t)kTok * kE * 4 + 65536 * sizeof(_Float16));
    float*    mask_sum  = probs_sum + kB * kE;

    float* dispatch = out;
    float* combine  = out + kDispatchElems;
    float* auxp     = out + 2 * kDispatchElems;

    zero_f4<<<4096, 256, 0, stream>>>((float4*)out, 2 * kDispatchElems / 4);
    zero_f<<<4, 256, 0, stream>>>(probs_sum, 2 * kB * kE);
    wconv<<<256, 256, 0, stream>>>(wg, wB);
    gating<<<kTok / 128, 256, 0, stream>>>(x, wB, weights, probs_sum, mask_sum);
    scan_scatter<<<kB * kE, 32, 0, stream>>>(weights, dispatch, combine);
    aux_kernel<<<1, 512, 0, stream>>>(probs_sum, mask_sum, auxp);
}
